// TransMIL_41214506172593
// MI455X (gfx1250) — compile-verified, matrix operand order checked by reading, not run on hardware
//
#include <hip/hip_runtime.h>
#include <cmath>

typedef __attribute__((ext_vector_type(16))) __bf16 v16bf;
typedef __attribute__((ext_vector_type(8)))  float  v8f;
typedef unsigned short u16;
typedef long long i64;

#define DIMC 512
#define NHEAD 8
#define DHEAD 64
#define LM 256
#define KCH 64   // K elements staged per LDS chunk

// ---------------- helpers ----------------
__device__ __forceinline__ float bf2f(u16 u) {
  union { unsigned int i; float f; } x; x.i = ((unsigned int)u) << 16; return x.f;
}
__device__ __forceinline__ u16 f2bf(float f) {
  union { float f; unsigned int i; } x; x.f = f;
  unsigned int u = x.i;
  unsigned int r = u + 0x7FFFu + ((u >> 16) & 1u);
  return (u16)(r >> 16);
}

// ---------------- WMMA GEMM (bf16 in, f32 accum) ----------------
// C[M,N] (+=) A[M,K] * B[K,N], B supplied transposed: Bt[N,K] row-major.
// Block = 256 threads = 8 waves. Block tile 256(M) x 32(N); wave tile 32x32
// (2x2 wmma accumulators). B chunk (32 cols x 64 k) staged into LDS with
// gfx1250 async-to-LDS loads, double buffered (ASYNCcnt + barrier).
// Requires: M%256==0, N%32==0, K%64==0, row strides multiple of 16 elems.
__global__ __launch_bounds__(256)
void gemm_bf16_wmma(const u16* __restrict__ A, i64 sA, int lda,
                    const int* __restrict__ rowmap, int rowoff,
                    const u16* __restrict__ Bt, i64 sB, int ldb,
                    float* __restrict__ C, i64 sC, int ldc,
                    int M, int N, int K,
                    const float* __restrict__ bias, int addC, int relu)
{
  __shared__ u16 shB[2][32 * KCH];          // 2 x 4 KB double buffer
  int tid  = threadIdx.x;
  int wave = tid >> 5;
  int lane = tid & 31;
  int n0 = blockIdx.x << 5;                 // 32 columns per block
  int m0 = (blockIdx.y << 8) + (wave << 5); // 32 rows per wave
  int b  = blockIdx.z;
  A  += (i64)b * sA;
  Bt += (i64)b * sB;
  C  += (i64)b * sC;
  int lr = lane & 15;
  int kh = lane & 16;                       // K half select (0 / 16)

  // A row bases for the two 16-row subtiles of this wave
  int ar0 = m0 + lr, ar1 = m0 + 16 + lr;
  i64 ao0 = rowmap ? (i64)(rowmap[ar0] + rowoff) * (i64)lda : (i64)ar0 * (i64)lda;
  i64 ao1 = rowmap ? (i64)(rowmap[ar1] + rowoff) * (i64)lda : (i64)ar1 * (i64)lda;
  const u16* ap0 = A + ao0 + kh;
  const u16* ap1 = A + ao1 + kh;

  // async B staging: thread tid copies 16 bytes of row (tid>>3)
  int brow = tid >> 3;                      // 0..31
  unsigned bvoff = (unsigned)(n0 + brow) * (unsigned)ldb * 2u + (unsigned)(tid & 7) * 16u;
  unsigned shb0 = (unsigned)(size_t)(&shB[0][0]);

  v8f acc00 = {0.f,0.f,0.f,0.f,0.f,0.f,0.f,0.f};
  v8f acc01 = acc00, acc10 = acc00, acc11 = acc00;

  // prologue: chunk 0 -> buffer 0
  {
    unsigned long long sa = (unsigned long long)(size_t)Bt;
    unsigned dst = shb0 + (unsigned)tid * 16u;
    asm volatile("global_load_async_to_lds_b128 %0, %1, %2"
                 :: "v"(dst), "v"(bvoff), "s"(sa) : "memory");
  }
  int cur = 0;
  for (int k = 0; k < K; k += KCH) {
    asm volatile("s_wait_asynccnt 0x0" ::: "memory");
    __syncthreads();                        // chunk[cur] resident for all waves
    if (k + KCH < K) {                      // overlap next chunk copy with compute
      unsigned long long sa = (unsigned long long)(size_t)Bt
                            + (unsigned long long)(k + KCH) * 2u;
      unsigned dst = shb0 + (unsigned)((cur ^ 1) * (32 * KCH * 2)) + (unsigned)tid * 16u;
      asm volatile("global_load_async_to_lds_b128 %0, %1, %2"
                   :: "v"(dst), "v"(bvoff), "s"(sa) : "memory");
    }
    __builtin_prefetch(ap0 + k + 4 * KCH, 0, 1);
    __builtin_prefetch(ap1 + k + 4 * KCH, 0, 1);
    const u16* shc = &shB[cur][0];
#pragma unroll
    for (int ks = 0; ks < KCH; ks += 32) {
      v16bf a0 = *(const v16bf*)(ap0 + k + ks);
      v16bf a1 = *(const v16bf*)(ap1 + k + ks);
      v16bf b0 = *(const v16bf*)(shc + lr * KCH + ks + kh);
      v16bf b1 = *(const v16bf*)(shc + (16 + lr) * KCH + ks + kh);
      acc00 = __builtin_amdgcn_wmma_f32_16x16x32_bf16(false, a0, false, b0, (short)0, acc00, false, false);
      acc01 = __builtin_amdgcn_wmma_f32_16x16x32_bf16(false, a0, false, b1, (short)0, acc01, false, false);
      acc10 = __builtin_amdgcn_wmma_f32_16x16x32_bf16(false, a1, false, b0, (short)0, acc10, false, false);
      acc11 = __builtin_amdgcn_wmma_f32_16x16x32_bf16(false, a1, false, b1, (short)0, acc11, false, false);
    }
    cur ^= 1;
  }

  // epilogue: 2x2 tiles, fused bias / beta-accumulate / relu
  int colA = n0 + lr, colB = n0 + 16 + lr;
  float biasA = bias ? bias[colA] : 0.0f;
  float biasB = bias ? bias[colB] : 0.0f;
  int rb0 = m0 + ((lane >> 4) << 3);
  int rb1 = m0 + 16 + ((lane >> 4) << 3);
#pragma unroll
  for (int r = 0; r < 8; ++r) {
    i64 c0 = (i64)(rb0 + r) * (i64)ldc;
    i64 c1 = (i64)(rb1 + r) * (i64)ldc;
    float v;
    v = acc00[r] + biasA; if (addC) v += C[c0 + colA]; if (relu) v = fmaxf(v, 0.f); C[c0 + colA] = v;
    v = acc01[r] + biasB; if (addC) v += C[c0 + colB]; if (relu) v = fmaxf(v, 0.f); C[c0 + colB] = v;
    v = acc10[r] + biasA; if (addC) v += C[c1 + colA]; if (relu) v = fmaxf(v, 0.f); C[c1 + colA] = v;
    v = acc11[r] + biasB; if (addC) v += C[c1 + colB]; if (relu) v = fmaxf(v, 0.f); C[c1 + colB] = v;
  }
}

// ---------------- elementwise / utility kernels ----------------
__global__ __launch_bounds__(256) void k_fill_u16(u16* d, u16 v, i64 n) {
  i64 i = (i64)blockIdx.x * 256 + threadIdx.x; if (i < n) d[i] = v;
}
__global__ __launch_bounds__(256) void k_fill_f32(float* d, float v, i64 n) {
  i64 i = (i64)blockIdx.x * 256 + threadIdx.x; if (i < n) d[i] = v;
}
__global__ __launch_bounds__(256) void k_conv_bf16(const float* s, u16* d, i64 n) {
  i64 i = (i64)blockIdx.x * 256 + threadIdx.x; if (i < n) d[i] = f2bf(s[i]);
}
__global__ __launch_bounds__(256) void k_copy_f32(float* d, const float* s, i64 n) {
  i64 i = (i64)blockIdx.x * 256 + threadIdx.x; if (i < n) d[i] = s[i];
}
// Bt[b][n*K+k] = W[b][k*N+n]  (convert + transpose)
__global__ __launch_bounds__(256) void k_convt_bf16(const float* W, u16* Bt,
                                                    int K, int N, i64 sW, i64 sBt) {
  i64 i = (i64)blockIdx.x * 256 + threadIdx.x;
  if (i >= (i64)K * N) return;
  int n = (int)(i / K); int k = (int)(i % K);
  Bt[(i64)blockIdx.y * sBt + i] = f2bf(W[(i64)blockIdx.y * sW + (i64)k * N + n]);
}
// w3 OIHW (512,512,3,3) -> w3t[tap][o][i]
__global__ __launch_bounds__(256) void k_w3t(const float* w3, u16* w3t) {
  i64 i = (i64)blockIdx.x * 256 + threadIdx.x;
  if (i >= 9LL * DIMC * DIMC) return;
  int tap = (int)(i / (DIMC * DIMC));
  int r = (int)(i % (DIMC * DIMC));
  int o = r / DIMC, ii = r % DIMC;
  w3t[i] = f2bf(w3[((i64)o * DIMC + ii) * 9 + tap]);
}

// layernorm over rows of 512, write bf16
__global__ __launch_bounds__(256) void k_layernorm_bf16(const float* X, const float* g,
                                                        const float* b, u16* out) {
  __shared__ float red[256];
  i64 row = blockIdx.x;
  const float* x = X + row * DIMC;
  int t = threadIdx.x;
  float x0 = x[t], x1 = x[t + 256];
  red[t] = x0 + x1; __syncthreads();
  for (int s = 128; s > 0; s >>= 1) { if (t < s) red[t] += red[t + s]; __syncthreads(); }
  float mu = red[0] * (1.0f / 512.0f); __syncthreads();
  float d0 = x0 - mu, d1 = x1 - mu;
  red[t] = d0 * d0 + d1 * d1; __syncthreads();
  for (int s = 128; s > 0; s >>= 1) { if (t < s) red[t] += red[t + s]; __syncthreads(); }
  float inv = rsqrtf(red[0] * (1.0f / 512.0f) + 1e-5f);
  out[row * DIMC + t]       = f2bf(d0 * inv * g[t] + b[t]);
  out[row * DIMC + t + 256] = f2bf(d1 * inv * g[t + 256] + b[t + 256]);
}

// split (npad,512) fp32 -> per-head bf16 [h][n][64], mask rows n<pad, scale
__global__ __launch_bounds__(256) void k_split_heads(const float* C, u16* dst,
                                                     int npad, int pad, float scale) {
  i64 i = (i64)blockIdx.x * 256 + threadIdx.x;
  i64 total = (i64)npad * DIMC;
  if (i >= total) return;
  int n = (int)(i / DIMC); int ch = (int)(i % DIMC);
  int h = ch >> 6, c = ch & 63;
  float v = (n < pad) ? 0.0f : C[i] * scale;
  dst[((i64)h * npad + n) * 64 + c] = f2bf(v);
}

// landmarks: dst[h][j][c] = sum_t src[h][j*l+t][c] / (mls_j + eps)
__global__ __launch_bounds__(256) void k_landmarks(const u16* src, u16* dst,
                                                   int npad, int l, int pad) {
  int i = blockIdx.x * 256 + threadIdx.x;
  if (i >= NHEAD * LM * DHEAD) return;
  int c = i & 63, j = (i >> 6) & 255, h = i >> 14;
  const u16* base = src + ((i64)h * npad + (i64)j * l) * 64 + c;
  float acc = 0.f;
  for (int t = 0; t < l; ++t) acc += bf2f(base[(i64)t * 64]);
  int mls = j * l + l - pad;
  if (mls < 0) mls = 0; if (mls > l) mls = l;
  dst[i] = f2bf(acc / ((float)mls + 1e-8f));
}

// vT[h][c][n] = vh[h][n][c]
__global__ __launch_bounds__(256) void k_transpose_hnc(const u16* vh, u16* vT, int npad) {
  i64 i = (i64)blockIdx.x * 256 + threadIdx.x;
  i64 total = (i64)NHEAD * npad * 64;
  if (i >= total) return;
  int n = (int)(i % npad); i64 r = i / npad;
  int c = (int)(r & 63); int h = (int)(r >> 6);
  vT[i] = vh[((i64)h * npad + n) * 64 + c];
}

// depthwise 33-tap conv over sequence: outh[h][n][c]
__global__ __launch_bounds__(256) void k_resconv(const u16* vh, const float* rw,
                                                 float* outh, int npad) {
  i64 i = (i64)blockIdx.x * 256 + threadIdx.x;
  i64 total = (i64)NHEAD * npad * 64;
  if (i >= total) return;
  int c = (int)(i & 63); i64 r = i >> 6;
  int n = (int)(r % npad); int h = (int)(r / npad);
  const u16* vb = vh + (i64)h * npad * 64 + c;
  const float* w = rw + h * 33;
  float acc = 0.f;
  for (int t = 0; t < 33; ++t) {
    int nn = n + t - 16;
    if (nn >= 0 && nn < npad) acc += w[t] * bf2f(vb[(i64)nn * 64]);
  }
  outh[i] = acc;
}

// outcat[n][h*64+c] = bf16(outh[h][n][c])
__global__ __launch_bounds__(256) void k_combine(const float* outh, u16* outcat, int npad) {
  i64 i = (i64)blockIdx.x * 256 + threadIdx.x;
  i64 total = (i64)npad * DIMC;
  if (i >= total) return;
  int n = (int)(i / DIMC); int ch = (int)(i % DIMC);
  int h = ch >> 6, c = ch & 63;
  outcat[i] = f2bf(outh[((i64)h * npad + n) * 64 + c]);
}

__global__ __launch_bounds__(256) void k_residual(float* X, const float* C,
                                                  int tokens, int pad) {
  i64 i = (i64)blockIdx.x * 256 + threadIdx.x;
  i64 total = (i64)tokens * DIMC;
  if (i >= total) return;
  X[i] += C[(i64)pad * DIMC + i];
}

// masked softmax over rows of 256
__global__ __launch_bounds__(256) void k_softmax256(const float* S, float* outF, u16* outB,
                                                    int rowsPerBatch, int rv0, int cv0) {
  __shared__ float red[256];
  i64 row = blockIdx.x;
  int nr = (int)(row % rowsPerBatch);
  int c = threadIdx.x;
  const float NEGF = -3.402823466e38f;
  float v = (nr >= rv0 && c >= cv0) ? S[row * 256 + c] : NEGF;
  red[c] = v; __syncthreads();
  for (int s = 128; s > 0; s >>= 1) { if (c < s) red[c] = fmaxf(red[c], red[c + s]); __syncthreads(); }
  float mx = red[0]; __syncthreads();
  float e = __expf(v - mx);
  red[c] = e; __syncthreads();
  for (int s = 128; s > 0; s >>= 1) { if (c < s) red[c] += red[c + s]; __syncthreads(); }
  float a = e / red[0];
  if (outF) outF[row * 256 + c] = a;
  if (outB) outB[row * 256 + c] = f2bf(a);
}

// masked softmax over wide rows (ncols), write bf16
__global__ __launch_bounds__(256) void k_softmax_wide(const float* S, u16* outB, int ncols,
                                                      int rowsPerBatch, int rv0, int colStart) {
  __shared__ float red[256];
  i64 row = blockIdx.x;
  int nr = (int)(row % rowsPerBatch);
  bool rvalid = nr >= rv0;
  const float NEGF = -3.402823466e38f;
  const float* Sr = S + row * (i64)ncols;
  int t = threadIdx.x;
  float mx = NEGF;
  for (int c = t; c < ncols; c += 256) {
    float v = (rvalid && c >= colStart) ? Sr[c] : NEGF;
    mx = fmaxf(mx, v);
  }
  red[t] = mx; __syncthreads();
  for (int s = 128; s > 0; s >>= 1) { if (t < s) red[t] = fmaxf(red[t], red[t + s]); __syncthreads(); }
  mx = red[0]; __syncthreads();
  float sum = 0.f;
  for (int c = t; c < ncols; c += 256) {
    float v = (rvalid && c >= colStart) ? Sr[c] : NEGF;
    sum += __expf(v - mx);
  }
  red[t] = sum; __syncthreads();
  for (int s = 128; s > 0; s >>= 1) { if (t < s) red[t] += red[t + s]; __syncthreads(); }
  float inv = 1.0f / red[0];
  u16* Or = outB + row * (i64)ncols;
  for (int c = t; c < ncols; c += 256) {
    float v = (rvalid && c >= colStart) ? Sr[c] : NEGF;
    Or[c] = f2bf(__expf(v - mx) * inv);
  }
}

// ---------------- pinv (fp32, 256x256 batched over heads) ----------------
__global__ __launch_bounds__(256) void k_gemm_f32(const float* A, const float* B, float* C,
                                                  int M, int N, int K,
                                                  i64 sA, i64 sB, i64 sC, float alpha) {
  i64 i = (i64)blockIdx.x * 256 + threadIdx.x;
  if (i >= (i64)M * N) return;
  int mm = (int)(i / N), nn = (int)(i % N);
  const float* a = A + (i64)blockIdx.y * sA + (i64)mm * K;
  const float* b = B + (i64)blockIdx.y * sB + nn;
  float acc = 0.f;
  for (int k = 0; k < K; ++k) acc += a[k] * b[(i64)k * N];
  C[(i64)blockIdx.y * sC + i] = alpha * acc;
}
__global__ __launch_bounds__(256) void k_eye_minus(float* D, const float* S, float cdiag) {
  i64 i = (i64)blockIdx.x * 256 + threadIdx.x;
  if (i >= (i64)NHEAD * LM * LM) return;
  int r = (int)((i >> 8) & 255), c = (int)(i & 255);
  D[i] = ((r == c) ? cdiag : 0.0f) - S[i];
}
__global__ void k_zero2(float* p) { if (threadIdx.x < 2) p[threadIdx.x] = 0.f; }
__global__ __launch_bounds__(256) void k_absmax_rc(const float* a2, float* s01) {
  __shared__ float red[256];
  int id = blockIdx.x;
  int isCol = id & 1; id >>= 1;
  int b = id >> 8, i = id & 255;
  const float* base = a2 + (i64)b * (LM * LM);
  int t = threadIdx.x;
  float v = fabsf(isCol ? base[(i64)t * LM + i] : base[(i64)i * LM + t]);
  red[t] = v; __syncthreads();
  for (int s = 128; s > 0; s >>= 1) { if (t < s) red[t] += red[t + s]; __syncthreads(); }
  if (t == 0) atomicMax((int*)(s01 + isCol), __float_as_int(red[0]));
}
__global__ __launch_bounds__(256) void k_make_z(const float* a2, float* z, const float* s01) {
  i64 i = (i64)blockIdx.x * 256 + threadIdx.x;
  if (i >= (i64)NHEAD * LM * LM) return;
  float sc = s01[0] * s01[1];
  int b = (int)(i >> 16), r = (int)((i >> 8) & 255), c = (int)(i & 255);
  z[i] = a2[((i64)b << 16) + ((i64)c << 8) + r] / sc;
}

// ---------------- PPEG kernels ----------------
__global__ __launch_bounds__(256) void k_ppeg_init(const float* X, const float* b7,
                                                   const float* b5, const float* b3,
                                                   float* fout, i64 total) {
  i64 i = (i64)blockIdx.x * 256 + threadIdx.x;
  if (i >= total) return;
  int c = (int)(i % DIMC);
  fout[i] = X[DIMC + i] + b7[c] + b5[c] + b3[c];
}
__global__ __launch_bounds__(256) void k_dwconv_add(const float* X, const float* w,
                                                    int ks, int pd, float* fout, int s) {
  i64 i = (i64)blockIdx.x * 256 + threadIdx.x;
  i64 total = (i64)s * s * DIMC;
  if (i >= total) return;
  int c = (int)(i % DIMC); i64 p = i / DIMC;
  int y = (int)(p / s), x = (int)(p % s);
  const float* wc = w + (i64)c * ks * ks;
  float acc = 0.f;
  for (int ky = 0; ky < ks; ++ky) {
    int yy = y + ky - pd; if (yy < 0 || yy >= s) continue;
    for (int kx = 0; kx < ks; ++kx) {
      int xx = x + kx - pd; if (xx < 0 || xx >= s) continue;
      acc += wc[ky * ks + kx] * X[((i64)1 + (i64)yy * s + xx) * DIMC + c];
    }
  }
  fout[i] += acc;
}
__global__ __launch_bounds__(256) void k_fpad(const float* X, u16* fpad, int s) {
  i64 i = (i64)blockIdx.x * 256 + threadIdx.x;
  i64 total = (i64)s * s * DIMC;
  if (i >= total) return;
  int c = (int)(i % DIMC); i64 p = i / DIMC;
  int y = (int)(p / s), x = (int)(p % s);
  fpad[(((i64)(y + 1) * (s + 2)) + (x + 1)) * DIMC + c] = f2bf(X[(i64)(1 + p) * DIMC + c]);
}
__global__ __launch_bounds__(256) void k_rowmap(int* rm, int s, int s2, i64 total) {
  i64 i = (i64)blockIdx.x * 256 + threadIdx.x;
  if (i < total) rm[i] = (int)((i / s) * s2 + (i % s));
}
__global__ __launch_bounds__(256) void k_copyback(float* X, const float* fout, i64 total) {
  i64 i = (i64)blockIdx.x * 256 + threadIdx.x;
  if (i < total) X[DIMC + i] = fout[i];
}
__global__ __launch_bounds__(256) void k_dup_rows(float* X, int N, i64 total) {
  i64 i = (i64)blockIdx.x * 256 + threadIdx.x;
  if (i < total) X[(i64)(1 + N) * DIMC + i] = X[DIMC + i];
}

// ---------------- final head ----------------
__global__ __launch_bounds__(256) void k_final(const float* X, const float* g, const float* b,
                                               const float* w2, const float* b2, float* out) {
  __shared__ float red[256];
  __shared__ float ln[DIMC];
  int t = threadIdx.x;
  float x0 = X[t], x1 = X[t + 256];
  red[t] = x0 + x1; __syncthreads();
  for (int s = 128; s > 0; s >>= 1) { if (t < s) red[t] += red[t + s]; __syncthreads(); }
  float mu = red[0] * (1.0f / 512.0f); __syncthreads();
  float d0 = x0 - mu, d1 = x1 - mu;
  red[t] = d0 * d0 + d1 * d1; __syncthreads();
  for (int s = 128; s > 0; s >>= 1) { if (t < s) red[t] += red[t + s]; __syncthreads(); }
  float inv = rsqrtf(red[0] * (1.0f / 512.0f) + 1e-5f);
  ln[t] = d0 * inv * g[t] + b[t];
  ln[t + 256] = d1 * inv * g[t + 256] + b[t + 256];
  __syncthreads();
  if (t < 2) {
    float acc = b2[t];
    for (int c = 0; c < DIMC; ++c) acc += ln[c] * w2[c * 2 + t];
    red[t] = acc;
  }
  __syncthreads();
  if (t == 0) {
    float l0 = red[0], l1 = red[1];
    out[0] = l0; out[1] = l1;
    out[2] = 1.0f / (1.0f + __expf(-l0));
    out[3] = 1.0f / (1.0f + __expf(-l1));
    out[4] = (l1 > l0) ? 1.0f : 0.0f;
  }
}

// ---------------- host ----------------
extern "C" void kernel_launch(void* const* d_in, const int* in_sizes, int n_in,
                              void* d_out, int out_size, void* d_ws, size_t ws_size,
                              hipStream_t stream)
{
  const float* h_in  = (const float*)d_in[0];
  const float* fc1_w = (const float*)d_in[1];
  const float* fc1_b = (const float*)d_in[2];
  const float* cls   = (const float*)d_in[3];
  const float* w7    = (const float*)d_in[4];
  const float* b7    = (const float*)d_in[5];
  const float* w5    = (const float*)d_in[6];
  const float* b5    = (const float*)d_in[7];
  const float* w3    = (const float*)d_in[8];
  const float* b3    = (const float*)d_in[9];
  const float* normg = (const float*)d_in[10];
  const float* normb = (const float*)d_in[11];
  const float* fc2w  = (const float*)d_in[12];
  const float* fc2b  = (const float*)d_in[13];
  const float* l_ng[2]  = {(const float*)d_in[14], (const float*)d_in[20]};
  const float* l_nb[2]  = {(const float*)d_in[15], (const float*)d_in[21]};
  const float* l_qkv[2] = {(const float*)d_in[16], (const float*)d_in[22]};
  const float* l_ow[2]  = {(const float*)d_in[17], (const float*)d_in[23]};
  const float* l_ob[2]  = {(const float*)d_in[18], (const float*)d_in[24]};
  const float* l_rw[2]  = {(const float*)d_in[19], (const float*)d_in[25]};

  const int FEA = in_sizes[1] / DIMC;          // 1024
  const int N   = in_sizes[0] / FEA;           // 25600
  int s = (int)ceil(sqrt((double)N));          // 160
  int add = s * s - N;                         // 0 here
  int tokens = N + add + 1;                    // 25601
  int pad = (LM - tokens % LM) % LM;           // 255
  int npad = tokens + pad;                     // 25856
  int l = npad / LM;                           // 101
  int cv0 = (l > 0) ? pad / l : 0;             // first valid landmark = 2
  int s2 = s + 2;
  i64 nfeat = (i64)s * s;

  // ---- bump allocator over d_ws ----
  char* base = (char*)d_ws;
  size_t off = 0;
  auto alloc = [&](size_t bytes) -> char* {
    char* p = base + off;
    off = (off + bytes + 255) & ~(size_t)255;
    return p;
  };
  float* X       = (float*)alloc((i64)tokens * DIMC * 4);
  u16* fc1wt     = (u16*)alloc((i64)FEA * DIMC * 2);
  u16* qkvwt[2]  = {(u16*)alloc((i64)3 * DIMC * DIMC * 2), (u16*)alloc((i64)3 * DIMC * DIMC * 2)};
  u16* outwt[2]  = {(u16*)alloc((i64)DIMC * DIMC * 2),     (u16*)alloc((i64)DIMC * DIMC * 2)};
  u16* w3t       = (u16*)alloc((i64)9 * DIMC * DIMC * 2);
  i64 hb_bytes   = (i64)N * FEA * 2;
  i64 outh_bytes = (i64)NHEAD * npad * 64 * 4;
  char* hb_outh  = alloc((size_t)(hb_bytes > outh_bytes ? hb_bytes : outh_bytes));
  i64 cb1 = (i64)npad * DIMC, cb2 = nfeat * DIMC;
  float* Cbuf    = (float*)alloc((size_t)((cb1 > cb2 ? cb1 : cb2) * 4));
  i64 xp1 = (i64)npad * DIMC, xp2 = (i64)s2 * s2 * DIMC;
  u16* xpadb     = (u16*)alloc((size_t)((xp1 > xp2 ? xp1 : xp2) * 2));
  u16* qh        = (u16*)alloc((i64)NHEAD * npad * 64 * 2);
  u16* kh_       = (u16*)alloc((i64)NHEAD * npad * 64 * 2);
  u16* vh        = (u16*)alloc((i64)NHEAD * npad * 64 * 2);
  u16* vT        = (u16*)alloc((i64)NHEAD * 64 * npad * 2);
  float* simbuf  = (float*)alloc((i64)NHEAD * npad * LM * 4);
  u16* abuf      = (u16*)alloc((i64)NHEAD * npad * LM * 2);
  u16* qlb       = (u16*)alloc((i64)NHEAD * LM * 64 * 2);
  u16* klb       = (u16*)alloc((i64)NHEAD * LM * 64 * 2);
  float* a2      = (float*)alloc((i64)NHEAD * LM * LM * 4);
  float* zb      = (float*)alloc((i64)NHEAD * LM * LM * 4);
  float* z2b     = (float*)alloc((i64)NHEAD * LM * LM * 4);
  float* xzb     = (float*)alloc((i64)NHEAD * LM * LM * 4);
  float* tb      = (float*)alloc((i64)NHEAD * LM * LM * 4);
  float* B3      = (float*)alloc((i64)NHEAD * LM * 64 * 4);
  float* W2      = (float*)alloc((i64)NHEAD * LM * 64 * 4);
  u16* W2t       = (u16*)alloc((i64)NHEAD * 64 * LM * 2);
  float* s01     = (float*)alloc(256);
  int* rowmap    = (int*)alloc(nfeat * 4);
  (void)ws_size;

  auto g1 = [](i64 n) { return dim3((unsigned)((n + 255) / 256)); };
  // block tile 256(M) x 32(N); requires M%256==0, N%32==0, K%64==0 (all hold here)
  auto gemmB = [&](const u16* A, i64 sA, int lda, const int* rmap, int roff,
                   const u16* Bt, i64 sB, int ldb, float* C, i64 sC, int ldc,
                   int M, int Nn, int K, const float* bias, int addC, int relu, int batch) {
    dim3 grid(Nn / 32, (M + 255) / 256, batch);
    gemm_bf16_wmma<<<grid, 256, 0, stream>>>(A, sA, lda, rmap, roff, Bt, sB, ldb,
                                             C, sC, ldc, M, Nn, K, bias, addC, relu);
  };

  // ---- weight conversions ----
  u16* hb = (u16*)hb_outh;
  k_conv_bf16<<<g1((i64)N * FEA), 256, 0, stream>>>(h_in, hb, (i64)N * FEA);
  k_convt_bf16<<<dim3((unsigned)(((i64)FEA * DIMC + 255) / 256), 1), 256, 0, stream>>>(
      fc1_w, fc1wt, FEA, DIMC, 0, 0);
  for (int L = 0; L < 2; ++L) {
    k_convt_bf16<<<dim3((unsigned)(((i64)DIMC * 3 * DIMC + 255) / 256), 1), 256, 0, stream>>>(
        l_qkv[L], qkvwt[L], DIMC, 3 * DIMC, 0, 0);
    k_convt_bf16<<<dim3((unsigned)(((i64)DIMC * DIMC + 255) / 256), 1), 256, 0, stream>>>(
        l_ow[L], outwt[L], DIMC, DIMC, 0, 0);
  }
  k_w3t<<<g1(9LL * DIMC * DIMC), 256, 0, stream>>>(w3, w3t);

  // ---- fc1: X[1..N] = relu(h @ fc1_w + b) ----
  gemmB(hb, 0, FEA, nullptr, 0, fc1wt, 0, FEA, X + DIMC, 0, DIMC,
        N, DIMC, FEA, fc1_b, 0, 1, 1);
  k_copy_f32<<<g1(DIMC), 256, 0, stream>>>(X, cls, DIMC);
  if (add > 0) k_dup_rows<<<g1((i64)add * DIMC), 256, 0, stream>>>(X, N, (i64)add * DIMC);

  // ---- nystrom attention block ----
  auto attention = [&](int L) {
    float* outh = (float*)hb_outh;
    if (pad > 0) k_fill_u16<<<g1((i64)pad * DIMC), 256, 0, stream>>>(xpadb, 0, (i64)pad * DIMC);
    k_layernorm_bf16<<<tokens, 256, 0, stream>>>(X, l_ng[L], l_nb[L], xpadb + (i64)pad * DIMC);
    // qkv projections (3 GEMMs on column slices of qkv_w)
    for (int part = 0; part < 3; ++part) {
      gemmB(xpadb, 0, DIMC, nullptr, 0, qkvwt[L] + (i64)part * DIMC * DIMC, 0, DIMC,
            Cbuf, 0, DIMC, npad, DIMC, DIMC, nullptr, 0, 0, 1);
      u16* dst = (part == 0) ? qh : (part == 1) ? kh_ : vh;
      float sc = (part == 0) ? 0.125f : 1.0f;
      k_split_heads<<<g1((i64)npad * DIMC), 256, 0, stream>>>(Cbuf, dst, npad, pad, sc);
    }
    k_landmarks<<<g1((i64)NHEAD * LM * 64), 256, 0, stream>>>(qh, qlb, npad, l, pad);
    k_landmarks<<<g1((i64)NHEAD * LM * 64), 256, 0, stream>>>(kh_, klb, npad, l, pad);
    // sim2 = q_l @ k_l^T -> softmax -> a2 (fp32)
    gemmB(qlb, (i64)LM * 64, 64, nullptr, 0, klb, (i64)LM * 64, 64,
          xzb, (i64)LM * LM, LM, LM, LM, 64, nullptr, 0, 0, NHEAD);
    k_softmax256<<<NHEAD * LM, 256, 0, stream>>>(xzb, a2, nullptr, LM, cv0, cv0);
    // moore-penrose pinv of a2
    k_zero2<<<1, 32, 0, stream>>>(s01);
    k_absmax_rc<<<2 * NHEAD * LM, 256, 0, stream>>>(a2, s01);
    k_make_z<<<g1((i64)NHEAD * LM * LM), 256, 0, stream>>>(a2, zb, s01);
    float* zc = zb; float* z2c = z2b;
    i64 sq = (i64)LM * LM;
    dim3 gsq((unsigned)((sq + 255) / 256), NHEAD);
    for (int it = 0; it < 6; ++it) {
      k_gemm_f32<<<gsq, 256, 0, stream>>>(a2, zc, xzb, LM, LM, LM, sq, sq, sq, 1.0f);
      k_eye_minus<<<g1((i64)NHEAD * sq), 256, 0, stream>>>(tb, xzb, 7.0f);
      k_gemm_f32<<<gsq, 256, 0, stream>>>(xzb, tb, z2c, LM, LM, LM, sq, sq, sq, 1.0f);
      k_eye_minus<<<g1((i64)NHEAD * sq), 256, 0, stream>>>(z2c, z2c, 15.0f);
      k_gemm_f32<<<gsq, 256, 0, stream>>>(xzb, z2c, tb, LM, LM, LM, sq, sq, sq, 1.0f);
      k_eye_minus<<<g1((i64)NHEAD * sq), 256, 0, stream>>>(tb, tb, 13.0f);
      k_gemm_f32<<<gsq, 256, 0, stream>>>(zc, tb, z2c, LM, LM, LM, sq, sq, sq, 0.25f);
      float* tmp = zc; zc = z2c; z2c = tmp;
    }
    // sim3 = q_l @ k^T -> softmax -> a3 (bf16) -> B3 = a3 @ v
    gemmB(qlb, (i64)LM * 64, 64, nullptr, 0, kh_, (i64)npad * 64, 64,
          simbuf, (i64)LM * npad, npad, LM, npad, 64, nullptr, 0, 0, NHEAD);
    k_softmax_wide<<<NHEAD * LM, 256, 0, stream>>>(simbuf, abuf, npad, LM, cv0, pad);
    k_transpose_hnc<<<g1((i64)NHEAD * npad * 64), 256, 0, stream>>>(vh, vT, npad);
    gemmB(abuf, (i64)LM * npad, npad, nullptr, 0, vT, (i64)64 * npad, npad,
          B3, (i64)LM * 64, 64, LM, 64, npad, nullptr, 0, 0, NHEAD);
    // W2 = pinv @ B3 (fp32) -> bf16 transposed
    dim3 gw((unsigned)(((i64)LM * 64 + 255) / 256), NHEAD);
    k_gemm_f32<<<gw, 256, 0, stream>>>(zc, B3, W2, LM, 64, LM,
                                       sq, (i64)LM * 64, (i64)LM * 64, 1.0f);
    k_convt_bf16<<<dim3((unsigned)(((i64)LM * 64 + 255) / 256), NHEAD), 256, 0, stream>>>(
        W2, W2t, LM, 64, (i64)LM * 64, (i64)64 * LM);
    // residual depthwise conv on v -> outh
    k_resconv<<<g1((i64)NHEAD * npad * 64), 256, 0, stream>>>(vh, l_rw[L], outh, npad);
    // sim1 = q @ k_l^T -> softmax -> a1 (bf16); outh += a1 @ W2
    gemmB(qh, (i64)npad * 64, 64, nullptr, 0, klb, (i64)LM * 64, 64,
          simbuf, (i64)npad * LM, LM, npad, LM, 64, nullptr, 0, 0, NHEAD);
    k_softmax256<<<(unsigned)((i64)NHEAD * npad), 256, 0, stream>>>(
        simbuf, nullptr, abuf, npad, pad, cv0);
    gemmB(abuf, (i64)npad * LM, LM, nullptr, 0, W2t, (i64)64 * LM, LM,
          outh, (i64)npad * 64, 64, npad, 64, LM, nullptr, 1, 0, NHEAD);
    // combine heads, out projection, residual into X
    k_combine<<<g1((i64)npad * DIMC), 256, 0, stream>>>(outh, xpadb, npad);
    gemmB(xpadb, 0, DIMC, nullptr, 0, outwt[L], 0, DIMC, Cbuf, 0, DIMC,
          npad, DIMC, DIMC, l_ob[L], 0, 0, 1);
    k_residual<<<g1((i64)tokens * DIMC), 256, 0, stream>>>(X, Cbuf, tokens, pad);
  };

  attention(0);

  // ---- PPEG ----
  {
    float* fout = Cbuf;
    i64 tot = nfeat * DIMC;
    k_ppeg_init<<<g1(tot), 256, 0, stream>>>(X, b7, b5, b3, fout, tot);
    k_dwconv_add<<<g1(tot), 256, 0, stream>>>(X, w7, 7, 3, fout, s);
    k_dwconv_add<<<g1(tot), 256, 0, stream>>>(X, w5, 5, 2, fout, s);
    u16* fpad = xpadb;
    i64 fp = (i64)s2 * s2 * DIMC;
    k_fill_u16<<<g1(fp), 256, 0, stream>>>(fpad, 0, fp);
    k_fpad<<<g1(tot), 256, 0, stream>>>(X, fpad, s);
    k_rowmap<<<g1(nfeat), 256, 0, stream>>>(rowmap, s, s2, nfeat);
    for (int tap = 0; tap < 9; ++tap) {
      int dy = tap / 3, dx = tap % 3;
      gemmB(fpad, 0, DIMC, rowmap, dy * s2 + dx, w3t + (i64)tap * DIMC * DIMC, 0, DIMC,
            fout, 0, DIMC, (int)nfeat, DIMC, DIMC, nullptr, 1, 0, 1);
    }
    k_copyback<<<g1(tot), 256, 0, stream>>>(X, fout, tot);
  }

  attention(1);

  // ---- final: ln(X)[0] @ fc2 + outputs ----
  float* out = (float*)d_out;
  k_final<<<1, 256, 0, stream>>>(X, normg, normb, fc2w, fc2b, out);
  k_fill_f32<<<g1((i64)2 * N), 256, 0, stream>>>(out + 5, 1.0f, (i64)2 * N);
}
